// MultiHeadAttentionBlock_43396349559394
// MI455X (gfx1250) — matrix operand order repaired and checked by reading, not compiled, not run on hardware
//
#include <hip/hip_runtime.h>

// ---------------------------------------------------------------------------
// MI455X (gfx1250, wave32) fused multi-head attention block.
// All GEMM-shaped math runs on v_wmma_f32_16x16x32_f16 (f16 in, f32 acc).
// Softmax statistics stay in fp32 (online / flash-attention style).
// K/V tiles and the f16 A-tiles are staged into LDS with the CDNA5 async
// copy path (global_load_async_to_lds_b128 + s_wait_asynccnt).
// ---------------------------------------------------------------------------

typedef __attribute__((ext_vector_type(8)))  _Float16 h8;
typedef __attribute__((ext_vector_type(16))) _Float16 h16;
typedef __attribute__((ext_vector_type(8)))  float    f8;

#define D_MODEL 1024
#define NH      16
#define DK      64
#define BB      2
#define SS      2048
#define MTOT    (BB * SS)        // 4096 rows in the flattened [B*S, D] GEMMs

// GEMM tiling: 256 threads = 8 waves; wave w owns a 16x64 strip of a 128x64 tile.
#define BM 128
#define BN 64
#define BK 32
#define LDA_PAD 40               // 32 + 8 halfs of padding per LDS row
#define LDB_PAD 40

// Truncate a flat pointer to the 32-bit LDS byte offset (LDS aperture keeps
// the offset in addr[31:0]; upper DWORD is the aperture id).
#define LDS_OFF(p) ((unsigned)(unsigned long long)(p))

__device__ __forceinline__ void async_b128(const void* gaddr, unsigned lds) {
  asm volatile("global_load_async_to_lds_b128 %0, %1, off"
               :: "v"(lds), "v"((unsigned long long)gaddr) : "memory");
}
__device__ __forceinline__ void wait_async0() {
  asm volatile("s_wait_asynccnt 0" ::: "memory");
}

__device__ __forceinline__ h16 cat8(h8 lo, h8 hi) {
  return __builtin_shufflevector(lo, hi, 0, 1, 2, 3, 4, 5, 6, 7,
                                         8, 9, 10, 11, 12, 13, 14, 15);
}

__device__ __forceinline__ f8 zero8() {
  f8 z;
#pragma unroll
  for (int i = 0; i < 8; ++i) z[i] = 0.0f;
  return z;
}

__device__ __forceinline__ f8 wmma_16x16x32(h16 a, h16 b, f8 c) {
  // (neg_a, A, neg_b, B, c_mod, C, reuse_a, reuse_b)
  return __builtin_amdgcn_wmma_f32_16x16x32_f16(false, a, false, b,
                                                (short)0, c, false, false);
}

__device__ __forceinline__ h8 cvt2h8(float4 a, float4 b) {
  h8 r;
  r[0] = (_Float16)a.x; r[1] = (_Float16)a.y;
  r[2] = (_Float16)a.z; r[3] = (_Float16)a.w;
  r[4] = (_Float16)b.x; r[5] = (_Float16)b.y;
  r[6] = (_Float16)b.z; r[7] = (_Float16)b.w;
  return r;
}

// ---------------------------------------------------------------------------
// Generic WMMA GEMM:  C[M,N] = A[M,K] * W[K,N],  M=4096, N=K=1024.
// OUT_MODE 0: C -> f16, head-split [B,H,S,64]             (Q / K projections)
// OUT_MODE 1: C -> f16, head-split transposed [B,H,64,S]  (V projection)
// OUT_MODE 2: C -> f32, row-major [M,N]                   (output projection)
// A_F16: A is f16 (context) -> staged with async copy (no conversion needed).
// ---------------------------------------------------------------------------
template <int OUT_MODE, bool A_F16>
__global__ __launch_bounds__(256)
void gemm_wmma(const void* __restrict__ Aptr, const float* __restrict__ W,
               void* __restrict__ Cptr) {
  __shared__ __attribute__((aligned(16))) _Float16 sA[BM * LDA_PAD];   // [m][k]
  __shared__ __attribute__((aligned(16))) _Float16 sBt[BN * LDB_PAD];  // [n][k]

  const int tid  = threadIdx.x;
  const int lane = tid & 31;
  const int wave = tid >> 5;
  const int hi   = lane >> 4;   // lane half selects K-chunk per WMMA layout
  const int l15  = lane & 15;

  const int blockN = blockIdx.x * BN;
  const int blockM = blockIdx.y * BM;

  f8 acc[4];
#pragma unroll
  for (int t = 0; t < 4; ++t) acc[t] = zero8();

  for (int k0 = 0; k0 < D_MODEL; k0 += BK) {
    // ---- stage A tile (128 x 32) into LDS as f16 -------------------------
    {
      const int row  = tid >> 1;          // 0..127
      const int half = (tid & 1) * 16;    // 0 / 16
      _Float16* dst = &sA[row * LDA_PAD + half];
      if (A_F16) {
        const _Float16* ga =
            (const _Float16*)Aptr + (size_t)(blockM + row) * D_MODEL + k0 + half;
        async_b128(ga,     LDS_OFF(dst));        // ASYNCcnt-tracked LDS fill
        async_b128(ga + 8, LDS_OFF(dst + 8));
      } else {
        const float* ga =
            (const float*)Aptr + (size_t)(blockM + row) * D_MODEL + k0 + half;
        float4 f0 = *(const float4*)(ga);
        float4 f1 = *(const float4*)(ga + 4);
        float4 f2 = *(const float4*)(ga + 8);
        float4 f3 = *(const float4*)(ga + 12);
        *(h8*)(dst)     = cvt2h8(f0, f1);
        *(h8*)(dst + 8) = cvt2h8(f2, f3);
      }
    }
    // ---- stage W tile (32 x 64) into LDS transposed [n][k] ---------------
    {
      const int kk = tid >> 3;            // 0..31
      const int n0 = (tid & 7) * 8;       // 0..56
      const float* gw = W + (size_t)(k0 + kk) * D_MODEL + blockN + n0;
      float4 f0 = *(const float4*)(gw);
      float4 f1 = *(const float4*)(gw + 4);
      h8 hv = cvt2h8(f0, f1);
#pragma unroll
      for (int j = 0; j < 8; ++j) sBt[(n0 + j) * LDB_PAD + kk] = hv[j];
      if (k0 + BK < D_MODEL) {
        __builtin_prefetch(gw + (size_t)BK * D_MODEL, 0, 1);  // global_prefetch_b8
      }
    }
    if (A_F16) wait_async0();
    __syncthreads();

    // ---- WMMA: one A fragment, four B fragments per wave -----------------
    const int arow = wave * 16 + l15;
    h16 afrag = cat8(*(const h8*)&sA[arow * LDA_PAD + hi * 16],
                     *(const h8*)&sA[arow * LDA_PAD + hi * 16 + 8]);
#pragma unroll
    for (int t = 0; t < 4; ++t) {
      const int col = t * 16 + l15;
      h16 bfrag = cat8(*(const h8*)&sBt[col * LDB_PAD + hi * 16],
                       *(const h8*)&sBt[col * LDB_PAD + hi * 16 + 8]);
      acc[t] = wmma_16x16x32(afrag, bfrag, acc[t]);
    }
    __syncthreads();
  }

  // ---- store: C layout is (M = r + 8*hi, N = l15) per 16x16 tile ---------
#pragma unroll
  for (int t = 0; t < 4; ++t) {
#pragma unroll
    for (int r = 0; r < 8; ++r) {
      const int grow = blockM + wave * 16 + r + hi * 8;
      const int gcol = blockN + t * 16 + l15;
      const float v  = acc[t][r];
      if (OUT_MODE == 2) {
        ((float*)Cptr)[(size_t)grow * D_MODEL + gcol] = v;
      } else {
        const int b  = grow >> 11;     // / 2048
        const int s  = grow & 2047;
        const int h  = gcol >> 6;
        const int dk = gcol & 63;
        const _Float16 hv = (_Float16)v;
        if (OUT_MODE == 0) {   // [B,H,S,64]
          ((_Float16*)Cptr)[(((size_t)(b * NH + h)) * SS + s) * DK + dk] = hv;
        } else {               // [B,H,64,S]
          ((_Float16*)Cptr)[(((size_t)(b * NH + h)) * DK + dk) * SS + s] = hv;
        }
      }
    }
  }
}

// ---------------------------------------------------------------------------
// Flash attention: one block = 128 q rows of one (b,h); 8 waves x 16 rows.
// Block-uniform k loop: K (32x64) and V^T (64x32) tiles are staged once per
// block into LDS with async copies and shared by all 8 waves. Waves past
// their causal bound skip compute but keep the barriers. The causal mask is
// applied only in the single diagonal-straddling block per wave.
// ---------------------------------------------------------------------------
__global__ __launch_bounds__(256)
void attn_wmma(const _Float16* __restrict__ Qh, const _Float16* __restrict__ Kh,
               const _Float16* __restrict__ Vt, _Float16* __restrict__ Ctx) {
  __shared__ __attribute__((aligned(16))) _Float16 sK[32 * 72];   // [krow][d]
  __shared__ __attribute__((aligned(16))) _Float16 sV[64 * 40];   // [dk][k]
  __shared__ __attribute__((aligned(16))) _Float16 sP[8][16 * 40];

  const int tid  = threadIdx.x;
  const int lane = tid & 31;
  const int wave = tid >> 5;
  const int hi   = lane >> 4;
  const int l15  = lane & 15;

  const int bh = blockIdx.y;        // b*NH + h
  const int b  = bh >> 4;
  const int h  = bh & 15;
  const int qb0   = blockIdx.x * 128;
  const int qbase = qb0 + wave * 16;

  const _Float16* Qb = Qh + (size_t)bh * SS * DK;
  const _Float16* Kb = Kh + (size_t)bh * SS * DK;
  const _Float16* Vb = Vt + (size_t)bh * DK * SS;

  // Q fragments for this wave's 16 rows: d 0..31 and d 32..63.
  const _Float16* qrow = Qb + (size_t)(qbase + l15) * DK;
  h16 aq0 = cat8(*(const h8*)(qrow + hi * 16),
                 *(const h8*)(qrow + hi * 16 + 8));
  h16 aq1 = cat8(*(const h8*)(qrow + 32 + hi * 16),
                 *(const h8*)(qrow + 40 + hi * 16));

  f8 o[4];
#pragma unroll
  for (int t = 0; t < 4; ++t) o[t] = zero8();
  float mrow[8], lrow[8];
#pragma unroll
  for (int r = 0; r < 8; ++r) { mrow[r] = -3.0e38f; lrow[r] = 0.0f; }

  const float scale = 0.125f;            // 1/sqrt(64)
  const int nkb_blk = (qb0 + 159) >> 5;  // block-uniform trip count
  const int nkb_wav = (qbase + 47) >> 5; // this wave's causal bound
  const int nfull   = (qbase + 1) >> 5;  // blocks fully below the diagonal

  for (int kb = 0; kb < nkb_blk; ++kb) {
    const int kcol0 = kb * 32;

    // ---- async-stage K and V^T tiles (4KB each) into LDS -----------------
    {
      const int krow = tid >> 3, kch = tid & 7;     // 32 rows x 8 chunks
      async_b128(Kb + (size_t)(kcol0 + krow) * DK + kch * 8,
                 LDS_OFF(&sK[krow * 72 + kch * 8]));
      const int vrow = tid >> 2, vch = tid & 3;     // 64 rows x 4 chunks
      async_b128(Vb + (size_t)vrow * SS + kcol0 + vch * 8,
                 LDS_OFF(&sV[vrow * 40 + vch * 8]));
      wait_async0();
    }
    __syncthreads();

    if (kb < nkb_wav) {
      // ---- scores: S = Q . K^T, two 16-col tiles, K-dim 64 = 2 WMMA each -
      f8 s0 = zero8(), s1 = zero8();
      {
        const _Float16* kr0 = &sK[(size_t)l15 * 72];
        const _Float16* kr1 = &sK[(size_t)(16 + l15) * 72];
        h16 bk00 = cat8(*(const h8*)(kr0 + hi * 16), *(const h8*)(kr0 + hi * 16 + 8));
        h16 bk01 = cat8(*(const h8*)(kr0 + 32 + hi * 16), *(const h8*)(kr0 + 40 + hi * 16));
        s0 = wmma_16x16x32(aq0, bk00, s0);
        s0 = wmma_16x16x32(aq1, bk01, s0);
        h16 bk10 = cat8(*(const h8*)(kr1 + hi * 16), *(const h8*)(kr1 + hi * 16 + 8));
        h16 bk11 = cat8(*(const h8*)(kr1 + 32 + hi * 16), *(const h8*)(kr1 + 40 + hi * 16));
        s1 = wmma_16x16x32(aq0, bk10, s1);
        s1 = wmma_16x16x32(aq1, bk11, s1);
      }

      // ---- scale + (diagonal-block-only) causal mask ---------------------
      float p0[8], p1[8];
#pragma unroll
      for (int r = 0; r < 8; ++r) {
        p0[r] = s0[r] * scale;
        p1[r] = s1[r] * scale;
      }
      if (kb >= nfull) {                 // wave-uniform branch: 1 block/wave
#pragma unroll
        for (int r = 0; r < 8; ++r) {
          const int row = qbase + r + hi * 8;
          if (kcol0 + l15 > row)      p0[r] = -1.0e30f;
          if (kcol0 + 16 + l15 > row) p1[r] = -1.0e30f;
        }
      }

      // ---- online softmax (fp32), row reductions across 16-lane halves ---
#pragma unroll
      for (int r = 0; r < 8; ++r) {
        float bm = fmaxf(p0[r], p1[r]);
#pragma unroll
        for (int x = 8; x >= 1; x >>= 1) bm = fmaxf(bm, __shfl_xor(bm, x, 32));
        const float mnew  = fmaxf(mrow[r], bm);
        const float alpha = __expf(mrow[r] - mnew);
        mrow[r] = mnew;
        const float e0 = __expf(p0[r] - mnew);
        const float e1 = __expf(p1[r] - mnew);
        p0[r] = e0; p1[r] = e1;
        float rs = e0 + e1;
#pragma unroll
        for (int x = 8; x >= 1; x >>= 1) rs += __shfl_xor(rs, x, 32);
        lrow[r] = lrow[r] * alpha + rs;
#pragma unroll
        for (int t = 0; t < 4; ++t) o[t][r] *= alpha;
      }

      // ---- relayout P: C-layout -> A-fragment via per-wave LDS patch -----
      _Float16* myP = sP[wave];
#pragma unroll
      for (int r = 0; r < 8; ++r) {
        const int prow = r + hi * 8;
        myP[prow * 40 + l15]      = (_Float16)p0[r];
        myP[prow * 40 + 16 + l15] = (_Float16)p1[r];
      }
      asm volatile("s_wait_dscnt 0" ::: "memory");  // same-wave DS ordering
      const _Float16* pr = &myP[l15 * 40];
      h16 ap = cat8(*(const h8*)(pr + hi * 16), *(const h8*)(pr + hi * 16 + 8));

      // ---- O += P @ V : V^T tile makes B-fragments contiguous ------------
#pragma unroll
      for (int t = 0; t < 4; ++t) {
        const _Float16* vr = &sV[(size_t)(t * 16 + l15) * 40 + hi * 16];
        h16 bv = cat8(*(const h8*)(vr), *(const h8*)(vr + 8));
        o[t] = wmma_16x16x32(ap, bv, o[t]);
      }
    }
    __syncthreads();
  }

  // ---- normalize and scatter context [B,S,D_MODEL] as f16 ----------------
#pragma unroll
  for (int r = 0; r < 8; ++r) {
    const float inv = 1.0f / lrow[r];
    const int srow  = qbase + r + hi * 8;
    _Float16* crow  = Ctx + ((size_t)b * SS + srow) * D_MODEL + h * DK;
#pragma unroll
    for (int t = 0; t < 4; ++t) crow[t * 16 + l15] = (_Float16)(o[t][r] * inv);
  }
}

// ---------------------------------------------------------------------------
// Launcher: 5 stream-ordered launches.
// inputs: q k v mask w_q w_k w_v w_o  (mask is causal -> computed analytically)
// workspace: Qh(8MB) Kh(8MB) Vt(8MB) Ctx(8MB) = 32MB of f16 intermediates.
// ---------------------------------------------------------------------------
extern "C" void kernel_launch(void* const* d_in, const int* in_sizes, int n_in,
                              void* d_out, int out_size, void* d_ws, size_t ws_size,
                              hipStream_t stream) {
  (void)in_sizes; (void)n_in; (void)out_size; (void)ws_size;

  const float* q  = (const float*)d_in[0];
  const float* k  = (const float*)d_in[1];
  const float* v  = (const float*)d_in[2];
  const float* wq = (const float*)d_in[4];
  const float* wk = (const float*)d_in[5];
  const float* wv = (const float*)d_in[6];
  const float* wo = (const float*)d_in[7];

  char* ws = (char*)d_ws;
  const size_t headBytes = (size_t)BB * NH * SS * DK * 2;  // 8 MB each
  _Float16* Qh  = (_Float16*)(ws);
  _Float16* Kh  = (_Float16*)(ws + headBytes);
  _Float16* Vt  = (_Float16*)(ws + 2 * headBytes);
  _Float16* Ctx = (_Float16*)(ws + 3 * headBytes);

  const dim3 blk(256);
  const dim3 ggrid(D_MODEL / BN, MTOT / BM);  // (16, 32)

  gemm_wmma<0, false><<<ggrid, blk, 0, stream>>>((const void*)q, wq, (void*)Qh);
  gemm_wmma<0, false><<<ggrid, blk, 0, stream>>>((const void*)k, wk, (void*)Kh);
  gemm_wmma<1, false><<<ggrid, blk, 0, stream>>>((const void*)v, wv, (void*)Vt);

  const dim3 agrid(SS / 128, BB * NH);        // (16, 32)
  attn_wmma<<<agrid, blk, 0, stream>>>(Qh, Kh, Vt, Ctx);

  gemm_wmma<2, true><<<ggrid, blk, 0, stream>>>((const void*)Ctx, wo, d_out);
}
